// KLFocalLossColBERT_74466142978305
// MI455X (gfx1250) — matrix operand order repaired and checked by reading, not compiled, not run on hardware
//
#include <hip/hip_runtime.h>
#include <hip/hip_bf16.h>
#include <math.h>

typedef _Float16 v16h __attribute__((ext_vector_type(16)));
typedef _Float16 v8h  __attribute__((ext_vector_type(8)));
typedef _Float16 v4h  __attribute__((ext_vector_type(4)));
typedef float    v8f  __attribute__((ext_vector_type(8)));

#define B_    128
#define LQ_   32
#define LD_   256
#define D_    128
#define NWAY_ 16
#define PITCH 136   // halfs per row in LDS tiles (272B: 16B-aligned, bank-spread)

__device__ __forceinline__ v16h cat16(v8h lo, v8h hi) {
  v16h r;
#pragma unroll
  for (int i = 0; i < 8; ++i) { r[i] = lo[i]; r[i + 8] = hi[i]; }
  return r;
}

// ---------------- Kernel 1: query L2-normalize (per token over D) ----------------
__global__ __launch_bounds__(256) void qnorm_kernel(const float* __restrict__ q,
                                                    float* __restrict__ qn) {
  const int tid  = threadIdx.x;
  const int lane = tid & 31;
  const int row  = blockIdx.x * 8 + (tid >> 5);   // 4096 rows total
  const float4* src = (const float4*)(q + (size_t)row * D_);
  float4 v = src[lane];
  float ss = v.x * v.x + v.y * v.y + v.z * v.z + v.w * v.w;
#pragma unroll
  for (int m = 16; m >= 1; m >>= 1) ss += __shfl_xor(ss, m, 32);
  float inv = 1.0f / fmaxf(sqrtf(ss), 1e-12f);
  float4 o; o.x = v.x * inv; o.y = v.y * inv; o.z = v.z * inv; o.w = v.w * inv;
  ((float4*)(qn + (size_t)row * D_))[lane] = o;
}

// ---------------- Kernel 2: per-(n,b) normalize-docs + WMMA GEMM + max/sum -------
__device__ __forceinline__ void gemm_half(const _Float16* __restrict__ qt,
                                          const _Float16* __restrict__ dt,
                                          int lane, int m, int c, v8f& vmax) {
  v8f acc0 = {0.f,0.f,0.f,0.f,0.f,0.f,0.f,0.f};
  v8f acc1 = acc0;
  const int aRowOff = (m * 16 + (lane & 15)) * PITCH + ((lane >> 4) << 4);
  const int bRowOff0 = (c * 32 + (lane & 15)) * PITCH + ((lane >> 4) << 4);
#pragma unroll
  for (int kt = 0; kt < 4; ++kt) {
    const int kk = kt * 32;
    // A fragment: lanes0-15 hold K kk..kk+15; lanes16-31 offset +16
    v8h a0 = *(const v8h*)(qt + aRowOff + kk);
    v8h a1 = *(const v8h*)(qt + aRowOff + kk + 8);
    v16h A = cat16(a0, a1);
    // B fragments: doc rows as columns; lanes0-15 K kk..kk+15, lanes16-31 +16
    v8h b0 = *(const v8h*)(dt + bRowOff0 + kk);
    v8h b1 = *(const v8h*)(dt + bRowOff0 + kk + 8);
    v16h B0 = cat16(b0, b1);
    acc0 = __builtin_amdgcn_wmma_f32_16x16x32_f16(false, A, false, B0,
                                                  (short)0, acc0, false, false);
    v8h b2 = *(const v8h*)(dt + bRowOff0 + 16 * PITCH + kk);
    v8h b3 = *(const v8h*)(dt + bRowOff0 + 16 * PITCH + kk + 8);
    v16h B1 = cat16(b2, b3);
    acc1 = __builtin_amdgcn_wmma_f32_16x16x32_f16(false, A, false, B1,
                                                  (short)0, acc1, false, false);
  }
#pragma unroll
  for (int i = 0; i < 8; ++i) vmax[i] = fmaxf(vmax[i], fmaxf(acc0[i], acc1[i]));
}

__global__ __launch_bounds__(256) void colbert_sim_kernel(
    const float* __restrict__ qn,        // (B, LQ, D) normalized f32
    const float* __restrict__ docs,      // (NWAY, B, LD, D)
    const int*   __restrict__ masks,     // (NWAY, B, LD)
    float* __restrict__ scores)          // (B, NWAY)
{
  __shared__ _Float16 dtile[128 * PITCH];   // half of doc tile, masked f16
  __shared__ _Float16 qtile[LQ_ * PITCH];   // Q' = qn * inv[dd]
  __shared__ float    red[256 * 4];
  __shared__ float    invn[D_];
  __shared__ float    msk[LD_];
  __shared__ float    rowmaxS[4 * LQ_];

  const int tid  = threadIdx.x;
  const int lane = tid & 31;
  const int bx   = blockIdx.x;            // bx = n*B + b
  const int b    = bx & (B_ - 1);
  const float* __restrict__ docp = docs + (size_t)bx * LD_ * D_;
  const float4* __restrict__ docp4 = (const float4*)docp;
  const int* __restrict__ mp = masks + (size_t)bx * LD_;

  msk[tid] = (float)mp[tid];
  if (tid < 128) msk[tid + 128] = (float)mp[tid + 128];
  __syncthreads();

  // ---- Pass 1: stream full tile, accumulate per-column sum-of-squares;
  //      stage masked f16 for rows 0..127 into LDS.
  const int cg = (tid & 31) * 4;          // starting column (4 per thread)
  const int rg = tid >> 5;                // row phase within group of 8
  float a0 = 0.f, a1 = 0.f, a2 = 0.f, a3 = 0.f;
#pragma unroll 4
  for (int it = 0; it < 16; ++it) {
    const int l = it * 8 + rg;
    float4 v = docp4[l * 32 + (tid & 31)];
    const float mv = msk[l];
    float x0 = v.x * mv, x1 = v.y * mv, x2 = v.z * mv, x3 = v.w * mv;
    a0 += x0 * x0; a1 += x1 * x1; a2 += x2 * x2; a3 += x3 * x3;
    v4h h = {(_Float16)x0, (_Float16)x1, (_Float16)x2, (_Float16)x3};
    *(v4h*)&dtile[l * PITCH + cg] = h;
  }
#pragma unroll 4
  for (int it = 16; it < 32; ++it) {
    const int l = it * 8 + rg;
    float4 v = docp4[l * 32 + (tid & 31)];
    const float mv = msk[l];
    float x0 = v.x * mv, x1 = v.y * mv, x2 = v.z * mv, x3 = v.w * mv;
    a0 += x0 * x0; a1 += x1 * x1; a2 += x2 * x2; a3 += x3 * x3;
  }
  red[tid * 4 + 0] = a0; red[tid * 4 + 1] = a1;
  red[tid * 4 + 2] = a2; red[tid * 4 + 3] = a3;
  __syncthreads();

  if (tid < 128) {                         // column dd = tid
    float s = 0.f;
#pragma unroll
    for (int g = 0; g < 8; ++g)
      s += red[(g * 32 + (tid >> 2)) * 4 + (tid & 3)];
    const float nrm = sqrtf(s);
    invn[tid] = (nrm > 1e-12f) ? (1.0f / nrm) : 0.0f;
  }
  __syncthreads();

  // ---- Build Q' = qn * inv[dd]  (fold doc column norm into query)
  const float* __restrict__ qrow = qn + (size_t)b * LQ_ * D_;
#pragma unroll
  for (int i = 0; i < 16; ++i) {
    const int idx = tid + 256 * i;
    const int r = idx >> 7, cidx = idx & 127;
    qtile[r * PITCH + cidx] = (_Float16)(qrow[idx] * invn[cidx]);
  }
  __syncthreads();

  // ---- GEMM + running max.  wave w: m = M-tile (q rows), c = 32-column chunk
  const int w = tid >> 5;
  const int m = w & 1;
  const int c = w >> 1;
  v8f vmax;
#pragma unroll
  for (int i = 0; i < 8; ++i) vmax[i] = -INFINITY;

  gemm_half(qtile, dtile, lane, m, c, vmax);   // doc rows 0..127
  __syncthreads();

  // restage rows 128..255 (hits L2, tile was just streamed)
#pragma unroll 4
  for (int it = 16; it < 32; ++it) {
    const int l = it * 8 + rg;
    float4 v = docp4[l * 32 + (tid & 31)];
    const float mv = msk[l];
    v4h h = {(_Float16)(v.x * mv), (_Float16)(v.y * mv),
             (_Float16)(v.z * mv), (_Float16)(v.w * mv)};
    *(v4h*)&dtile[(l - 128) * PITCH + cg] = h;
  }
  __syncthreads();
  gemm_half(qtile, dtile, lane, m, c, vmax);   // doc rows 128..255

  // ---- reduce max over N (lane groups of 16; C layout: N = lane%16)
#pragma unroll
  for (int i = 0; i < 8; ++i) {
    float v = vmax[i];
    v = fmaxf(v, __shfl_xor(v, 1, 32));
    v = fmaxf(v, __shfl_xor(v, 2, 32));
    v = fmaxf(v, __shfl_xor(v, 4, 32));
    v = fmaxf(v, __shfl_xor(v, 8, 32));
    vmax[i] = v;
  }
  if ((lane & 15) == 0) {
    const int h = lane >> 4;                   // C layout: M = vgpr + 8*(lane/16)
#pragma unroll
    for (int vi = 0; vi < 8; ++vi)
      rowmaxS[c * 32 + m * 16 + h * 8 + vi] = vmax[vi];
  }
  __syncthreads();

  if (tid < 32) {
    float rm = rowmaxS[tid];
#pragma unroll
    for (int cc = 1; cc < 4; ++cc) rm = fmaxf(rm, rowmaxS[cc * 32 + tid]);
#pragma unroll
    for (int mm = 16; mm >= 1; mm >>= 1) rm += __shfl_xor(rm, mm, 32);
    if (tid == 0) scores[(size_t)b * NWAY_ + (bx >> 7)] = rm;
  }
}

// ---------------- Kernel 3: log-softmax + focal KL + mean ----------------
__global__ __launch_bounds__(128) void loss_kernel(const float* __restrict__ scores,
                                                   const float* __restrict__ labels,
                                                   float* __restrict__ out) {
  __shared__ float sred[128];
  const int b = threadIdx.x;
  float s[NWAY_];
  float mx = -INFINITY;
#pragma unroll
  for (int j = 0; j < NWAY_; ++j) { s[j] = scores[b * NWAY_ + j]; mx = fmaxf(mx, s[j]); }
  float se = 0.f;
#pragma unroll
  for (int j = 0; j < NWAY_; ++j) se += expf(s[j] - mx);
  const float lse = mx + logf(se);
  float a = 0.f;
#pragma unroll
  for (int j = 0; j < NWAY_; ++j) {
    const float lp = s[j] - lse;
    const float p  = expf(lp);
    const float t  = labels[b * (2 * NWAY_) + j];
    const float kl = expf(t) * (t - lp);
    const float p2 = p * p;
    a += kl * p2 * p2 * p;                 // GAMMA = 5
  }
  sred[b] = a;
  __syncthreads();
#pragma unroll
  for (int st = 64; st >= 1; st >>= 1) {
    if (b < st) sred[b] += sred[b + st];
    __syncthreads();
  }
  if (b == 0) out[0] = sred[0] / (float)(B_ * NWAY_);
}

extern "C" void kernel_launch(void* const* d_in, const int* in_sizes, int n_in,
                              void* d_out, int out_size, void* d_ws, size_t ws_size,
                              hipStream_t stream) {
  const float* qreps  = (const float*)d_in[0];   // (B, LQ, D)
  const float* dreps  = (const float*)d_in[1];   // (NWAY, B, LD, D)
  const int*   masks  = (const int*)d_in[2];     // (NWAY, B, LD)
  const float* labels = (const float*)d_in[3];   // (B, 2*NWAY)
  float* out = (float*)d_out;

  float* qn     = (float*)d_ws;                  // B*LQ*D f32 = 2 MB
  float* scores = qn + (size_t)B_ * LQ_ * D_;    // B*NWAY f32

  qnorm_kernel<<<(B_ * LQ_) / 8, 256, 0, stream>>>(qreps, qn);
  colbert_sim_kernel<<<NWAY_ * B_, 256, 0, stream>>>(qn, dreps, masks, scores);
  loss_kernel<<<1, 128, 0, stream>>>(scores, labels, out);
}
